// SwinTransformerBlock_42880953484153
// MI455X (gfx1250) — matrix operand order repaired and checked by reading, not compiled, not run on hardware
//
#include <hip/hip_runtime.h>

#define TOKENS 50176      // B * H * W
#define CH     256
#define NH     8
#define HD     32
#define WSZ    7
#define WA     49
#define BNW    1024       // total windows (B * 64)
#define MLPD   1024
#define HWDIM  56

typedef __attribute__((ext_vector_type(16))) _Float16 v16h;
typedef __attribute__((ext_vector_type(8)))  _Float16 v8h;
typedef __attribute__((ext_vector_type(8)))  float    v8f;

// ---------------------------------------------------------------------------
// WMMA fragment loaders matching gfx1250 16-bit A/B VGPR striping.
// A (16x32, MxK): lanes 0-15 row=lane, elems 0..7 = K kb..kb+7, 8..15 = K kb+16..kb+23 (kb=0);
//                 lanes 16-31 same rows with kb = 8.
// B (32x16, KxN): lane(&15) = column N, elems 0..7 = K kb..kb+7, 8..15 = K kb+16..kb+23 (kb = 0 or 8 by lane half).
// ---------------------------------------------------------------------------
__device__ __forceinline__ v16h load_a_frag(const _Float16* p) {
    v8h lo = *(const v8h*)p;          // K kb .. kb+7
    v8h hi = *(const v8h*)(p + 16);   // K kb+16 .. kb+23
    return __builtin_shufflevector(lo, hi, 0,1,2,3,4,5,6,7,8,9,10,11,12,13,14,15);
}
__device__ __forceinline__ v16h load_b_frag(const _Float16* p) {
    v8h lo = *(const v8h*)p;
    v8h hi = *(const v8h*)(p + 16);
    return __builtin_shufflevector(lo, hi, 0,1,2,3,4,5,6,7,8,9,10,11,12,13,14,15);
}
__device__ __forceinline__ v8f wmma_f16(v16h a, v16h b, v8f c) {
    return __builtin_amdgcn_wmma_f32_16x16x32_f16(false, a, false, b, (short)0, c, false, false);
}

// ---------------------------------------------------------------------------
// Register-tiled GEMM core: one wave computes a 32x64 output tile (2M x 4N
// 16x16 WMMA tiles). A [M,K] row-major f16, Bt [N,K] row-major f16.
// 12 b128 loads per 8 wmma (A frags reused 4x, B frags reused 2x).
// ---------------------------------------------------------------------------
__device__ __forceinline__ void gemm_tile_2x4(const _Float16* __restrict__ A,
                                              const _Float16* __restrict__ Bt,
                                              int K, int row0, int col0, int lane,
                                              v8f acc[2][4]) {
    const _Float16* ap0 = A  + (size_t)(row0 + (lane & 15)) * K + ((lane >> 4) << 3);
    const _Float16* ap1 = ap0 + (size_t)16 * K;
    const _Float16* bp  = Bt + (size_t)(col0 + (lane & 15)) * K + ((lane >> 4) << 3);
    for (int k0 = 0; k0 < K; k0 += 32) {
        v16h a0 = load_a_frag(ap0 + k0);
        v16h a1 = load_a_frag(ap1 + k0);
        v16h b0 = load_b_frag(bp + k0);
        v16h b1 = load_b_frag(bp + (size_t)16 * K + k0);
        v16h b2 = load_b_frag(bp + (size_t)32 * K + k0);
        v16h b3 = load_b_frag(bp + (size_t)48 * K + k0);
        acc[0][0] = wmma_f16(a0, b0, acc[0][0]);
        acc[0][1] = wmma_f16(a0, b1, acc[0][1]);
        acc[0][2] = wmma_f16(a0, b2, acc[0][2]);
        acc[0][3] = wmma_f16(a0, b3, acc[0][3]);
        acc[1][0] = wmma_f16(a1, b0, acc[1][0]);
        acc[1][1] = wmma_f16(a1, b1, acc[1][1]);
        acc[1][2] = wmma_f16(a1, b2, acc[1][2]);
        acc[1][3] = wmma_f16(a1, b3, acc[1][3]);
    }
}

// ---------------------------------------------------------------------------
// Weight convert+transpose: W fp32 [K,N] -> Wt f16 [N,K]
// ---------------------------------------------------------------------------
__global__ __launch_bounds__(256) void cvt_wt_kernel(const float* __restrict__ W,
                                                     _Float16* __restrict__ Wt,
                                                     int K, int N) {
    int idx = blockIdx.x * 256 + threadIdx.x;
    if (idx < K * N) {
        int n = idx / K, k = idx - n * K;
        Wt[idx] = (_Float16)W[(size_t)k * N + n];
    }
}

// ---------------------------------------------------------------------------
// LayerNorm (+ optional cyclic shift & window partition). One wave per token.
// ---------------------------------------------------------------------------
__global__ __launch_bounds__(256) void ln_win_kernel(const float* __restrict__ x,
                                                     const float* __restrict__ g,
                                                     const float* __restrict__ be,
                                                     _Float16* __restrict__ h16, int mode) {
    int t    = blockIdx.x * 8 + (threadIdx.x >> 5);
    int lane = threadIdx.x & 31;
    size_t src;
    if (mode) {
        int b = t / 3136, rr = t - b * 3136;
        int w = rr / WA,  p  = rr - WA * w;
        int wy = w >> 3, wx = w & 7;
        int iy = p / WSZ, ix = p - WSZ * iy;
        int ys = wy * WSZ + iy, xs = wx * WSZ + ix;
        int yy = ys + 3; if (yy >= HWDIM) yy -= HWDIM;   // roll(-3)
        int xx = xs + 3; if (xx >= HWDIM) xx -= HWDIM;
        src = ((size_t)b * 3136 + yy * HWDIM + xx) * CH;
    } else {
        src = (size_t)t * CH;
    }
    float v[8]; float s = 0.f;
    #pragma unroll
    for (int j = 0; j < 8; j++) { v[j] = x[src + lane + 32 * j]; s += v[j]; }
    #pragma unroll
    for (int d = 1; d < 32; d <<= 1) s += __shfl_xor(s, d, 32);
    float mean = s * (1.0f / CH);
    float vs = 0.f;
    #pragma unroll
    for (int j = 0; j < 8; j++) { float dd = v[j] - mean; vs += dd * dd; }
    #pragma unroll
    for (int d = 1; d < 32; d <<= 1) vs += __shfl_xor(vs, d, 32);
    float inv = rsqrtf(vs * (1.0f / CH) + 1e-5f);
    #pragma unroll
    for (int j = 0; j < 8; j++) {
        int c = lane + 32 * j;
        h16[(size_t)t * CH + c] = (_Float16)((v[j] - mean) * inv * g[c] + be[c]);
    }
}

// ---------------------------------------------------------------------------
// Zero the WA..63 padding of q16/k16 (rows) and vT16 (columns).
// ---------------------------------------------------------------------------
__global__ __launch_bounds__(256) void pad_zero_kernel(_Float16* q16, _Float16* k16,
                                                       _Float16* vT16) {
    int idx = blockIdx.x * 256 + threadIdx.x;             // 8192 * 15 * 32
    if (idx >= 8192 * 15 * 32) return;
    int n  = idx & 31;
    int r  = (idx >> 5) % 15;
    int wh = idx / (15 * 32);
    q16 [((size_t)wh * 64 + WA + r) * 32 + n] = (_Float16)0.f;
    k16 [((size_t)wh * 64 + WA + r) * 32 + n] = (_Float16)0.f;
    vT16[((size_t)wh * 32 + n) * 64 + WA + r] = (_Float16)0.f;
}

// ---------------------------------------------------------------------------
// QKV GEMM: [50176,256] x [256,768] ; scatter to per-(window,head) Q/K/V^T f16.
// ---------------------------------------------------------------------------
__global__ __launch_bounds__(256) void gemm_qkv_kernel(const _Float16* __restrict__ A,
                                                       const _Float16* __restrict__ Bt,
                                                       const float* __restrict__ bias,
                                                       _Float16* __restrict__ q16,
                                                       _Float16* __restrict__ k16,
                                                       _Float16* __restrict__ vT16) {
    int lane = threadIdx.x & 31;
    int row0 = (blockIdx.y * 8 + (threadIdx.x >> 5)) * 32;
    int col0 = blockIdx.x * 64;
    v8f acc[2][4] = {};
    gemm_tile_2x4(A, Bt, CH, row0, col0, lane, acc);
    int cj = lane & 15, half = lane >> 4;
    #pragma unroll
    for (int ni = 0; ni < 4; ni++) {
        int f = col0 + ni * 16 + cj;
        float bv = bias[f];
        int part = f >> 8, fh = f & 255, hh = fh >> 5, n = fh & 31;
        #pragma unroll
        for (int mi = 0; mi < 2; mi++) {
            #pragma unroll
            for (int r = 0; r < 8; r++) {
                int t = row0 + mi * 16 + r + 8 * half;
                int w = t / WA, p = t - WA * w;
                int wh = w * NH + hh;
                float v = acc[mi][ni][r] + bv;
                if (part == 0)
                    q16[((size_t)wh * 64 + p) * 32 + n] =
                        (_Float16)(v * 0.17677669529663687f);   // HD^-0.5
                else if (part == 1)
                    k16[((size_t)wh * 64 + p) * 32 + n] = (_Float16)v;
                else
                    vT16[((size_t)wh * 32 + n) * 64 + p] = (_Float16)v;
            }
        }
    }
}

// ---------------------------------------------------------------------------
// Windowed attention: 1 block per window, 1 wave per head.
// QK^T (16 wmma), bias+mask+softmax in-register, P staged through LDS to
// convert C-layout -> A-layout, P*V (16 wmma).
// ---------------------------------------------------------------------------
__global__ __launch_bounds__(256) void attn_kernel(const _Float16* __restrict__ q16,
                                                   const _Float16* __restrict__ k16,
                                                   const _Float16* __restrict__ vT16,
                                                   const float* __restrict__ rpt,
                                                   _Float16* __restrict__ attnout) {
    __shared__ __align__(16) _Float16 P[NH][64 * 64];
    int w    = blockIdx.x;
    int h    = threadIdx.x >> 5;
    int lane = threadIdx.x & 31;
    int wh   = w * NH + h;
    const _Float16* qp = q16  + (size_t)wh * 64 * 32;
    const _Float16* kp = k16  + (size_t)wh * 64 * 32;
    const _Float16* vp = vT16 + (size_t)wh * 32 * 64;

    int arow = lane & 15;
    int akb  = (lane >> 4) << 3;   // A half-offset: 0 / 8
    int bkb  = (lane >> 4) << 3;   // B half-offset: 0 / 8
    int cj   = lane & 15, half = lane >> 4;

    // ---- scores = (Q*scale) @ K^T : K-dim = HD = 32, one wmma per tile
    v16h qa[4], kb[4];
    #pragma unroll
    for (int mi = 0; mi < 4; mi++) qa[mi] = load_a_frag(qp + (mi * 16 + arow) * 32 + akb);
    #pragma unroll
    for (int ni = 0; ni < 4; ni++) kb[ni] = load_b_frag(kp + (ni * 16 + arow) * 32 + bkb);
    v8f s[4][4];
    #pragma unroll
    for (int mi = 0; mi < 4; mi++)
        #pragma unroll
        for (int ni = 0; ni < 4; ni++) {
            v8f z = {};
            s[mi][ni] = wmma_f16(qa[mi], kb[ni], z);
        }

    // ---- relative-position bias + shifted-window mask + padding
    int wl = w & 63, wy = wl >> 3, wx = wl & 7;
    #pragma unroll
    for (int mi = 0; mi < 4; mi++) {
        #pragma unroll
        for (int r = 0; r < 8; r++) {
            int i = mi * 16 + 8 * half + r;
            int yi = i / WSZ, xi = i - WSZ * yi;
            int gyi = wy * WSZ + yi, gxi = wx * WSZ + xi;
            int regi = (gyi < 49 ? 0 : (gyi < 53 ? 1 : 2)) * 3
                     + (gxi < 49 ? 0 : (gxi < 53 ? 1 : 2));
            #pragma unroll
            for (int ni = 0; ni < 4; ni++) {
                int j = ni * 16 + cj;
                float v = s[mi][ni][r];
                if (i < WA && j < WA) {
                    int yj = j / WSZ, xj = j - WSZ * yj;
                    int ridx = (yi - yj + 6) * 13 + (xi - xj + 6);
                    v += rpt[ridx * NH + h];
                    int gyj = wy * WSZ + yj, gxj = wx * WSZ + xj;
                    int regj = (gyj < 49 ? 0 : (gyj < 53 ? 1 : 2)) * 3
                             + (gxj < 49 ? 0 : (gxj < 53 ? 1 : 2));
                    if (regi != regj) v -= 100.0f;
                } else {
                    v = -1e30f;
                }
                s[mi][ni][r] = v;
            }
        }
    }

    // ---- softmax per row (16-lane xor reductions stay within each half)
    _Float16* pl = P[h];
    #pragma unroll
    for (int mi = 0; mi < 4; mi++) {
        #pragma unroll
        for (int r = 0; r < 8; r++) {
            float mx = -3e38f;
            #pragma unroll
            for (int ni = 0; ni < 4; ni++) mx = fmaxf(mx, s[mi][ni][r]);
            #pragma unroll
            for (int d = 1; d < 16; d <<= 1) mx = fmaxf(mx, __shfl_xor(mx, d, 32));
            float sum = 0.f;
            #pragma unroll
            for (int ni = 0; ni < 4; ni++) {
                float e = __expf(s[mi][ni][r] - mx);
                s[mi][ni][r] = e; sum += e;
            }
            #pragma unroll
            for (int d = 1; d < 16; d <<= 1) sum += __shfl_xor(sum, d, 32);
            float inv = 1.0f / sum;
            int i = mi * 16 + 8 * half + r;
            #pragma unroll
            for (int ni = 0; ni < 4; ni++)
                pl[i * 64 + ni * 16 + cj] = (_Float16)(s[mi][ni][r] * inv);
        }
    }
    asm volatile("s_wait_dscnt 0x0" ::: "memory");   // LDS RAW fence (per-wave)

    // ---- out = P @ V : K-dim = 64 (padded), two k-steps of 32
    v8f o[4][2] = {};
    #pragma unroll
    for (int ks = 0; ks < 2; ks++) {
        v16h vb[2];
        #pragma unroll
        for (int nj = 0; nj < 2; nj++)
            vb[nj] = load_b_frag(vp + (nj * 16 + cj) * 64 + ks * 32 + bkb);
        #pragma unroll
        for (int mi = 0; mi < 4; mi++) {
            v16h pa = load_a_frag(pl + (mi * 16 + arow) * 64 + ks * 32 + akb);
            #pragma unroll
            for (int nj = 0; nj < 2; nj++)
                o[mi][nj] = wmma_f16(pa, vb[nj], o[mi][nj]);
        }
    }
    #pragma unroll
    for (int mi = 0; mi < 4; mi++)
        #pragma unroll
        for (int nj = 0; nj < 2; nj++)
            #pragma unroll
            for (int r = 0; r < 8; r++) {
                int i = mi * 16 + 8 * half + r;
                if (i < WA)
                    attnout[((size_t)w * WA + i) * CH + h * HD + nj * 16 + cj] =
                        (_Float16)o[mi][nj][r];
            }
}

// ---------------------------------------------------------------------------
// Proj GEMM + window reverse + un-shift + residual (fp32 ybuf)
// ---------------------------------------------------------------------------
__global__ __launch_bounds__(256) void gemm_proj_kernel(const _Float16* __restrict__ A,
                                                        const _Float16* __restrict__ Bt,
                                                        const float* __restrict__ bias,
                                                        const float* __restrict__ x0,
                                                        float* __restrict__ ybuf) {
    int lane = threadIdx.x & 31;
    int row0 = (blockIdx.y * 8 + (threadIdx.x >> 5)) * 32;
    int col0 = blockIdx.x * 64;
    v8f acc[2][4] = {};
    gemm_tile_2x4(A, Bt, CH, row0, col0, lane, acc);
    int cj = lane & 15, half = lane >> 4;
    #pragma unroll
    for (int ni = 0; ni < 4; ni++) {
        int c = col0 + ni * 16 + cj;
        float bv = bias[c];
        #pragma unroll
        for (int mi = 0; mi < 2; mi++) {
            #pragma unroll
            for (int r = 0; r < 8; r++) {
                int t = row0 + mi * 16 + r + 8 * half;
                int b = t / 3136, rr = t - b * 3136;
                int w = rr / WA,  p  = rr - WA * w;
                int wy = w >> 3, wx = w & 7;
                int iy = p / WSZ, ix = p - WSZ * iy;
                int ys = wy * WSZ + iy, xs = wx * WSZ + ix;
                int yy = ys + 3; if (yy >= HWDIM) yy -= HWDIM;   // roll(+3)
                int xx = xs + 3; if (xx >= HWDIM) xx -= HWDIM;
                size_t o = ((size_t)b * 3136 + yy * HWDIM + xx) * CH + c;
                ybuf[o] = x0[o] + acc[mi][ni][r] + bv;
            }
        }
    }
}

// ---------------------------------------------------------------------------
// FC1 GEMM + exact GELU -> f16 activations
// ---------------------------------------------------------------------------
__global__ __launch_bounds__(256) void gemm_fc1_kernel(const _Float16* __restrict__ A,
                                                       const _Float16* __restrict__ Bt,
                                                       const float* __restrict__ bias,
                                                       _Float16* __restrict__ act) {
    int lane = threadIdx.x & 31;
    int row0 = (blockIdx.y * 8 + (threadIdx.x >> 5)) * 32;
    int col0 = blockIdx.x * 64;
    v8f acc[2][4] = {};
    gemm_tile_2x4(A, Bt, CH, row0, col0, lane, acc);
    int cj = lane & 15, half = lane >> 4;
    #pragma unroll
    for (int ni = 0; ni < 4; ni++) {
        int c = col0 + ni * 16 + cj;
        float bv = bias[c];
        #pragma unroll
        for (int mi = 0; mi < 2; mi++) {
            #pragma unroll
            for (int r = 0; r < 8; r++) {
                int t = row0 + mi * 16 + r + 8 * half;
                float v = acc[mi][ni][r] + bv;
                float g = 0.5f * v * (1.0f + erff(v * 0.7071067811865476f));
                act[(size_t)t * MLPD + c] = (_Float16)g;
            }
        }
    }
}

// ---------------------------------------------------------------------------
// FC2 GEMM + residual -> final fp32 output
// ---------------------------------------------------------------------------
__global__ __launch_bounds__(256) void gemm_fc2_kernel(const _Float16* __restrict__ A,
                                                       const _Float16* __restrict__ Bt,
                                                       const float* __restrict__ bias,
                                                       const float* __restrict__ ybuf,
                                                       float* __restrict__ out) {
    int lane = threadIdx.x & 31;
    int row0 = (blockIdx.y * 8 + (threadIdx.x >> 5)) * 32;
    int col0 = blockIdx.x * 64;
    v8f acc[2][4] = {};
    gemm_tile_2x4(A, Bt, MLPD, row0, col0, lane, acc);
    int cj = lane & 15, half = lane >> 4;
    #pragma unroll
    for (int ni = 0; ni < 4; ni++) {
        int c = col0 + ni * 16 + cj;
        float bv = bias[c];
        #pragma unroll
        for (int mi = 0; mi < 2; mi++) {
            #pragma unroll
            for (int r = 0; r < 8; r++) {
                int t = row0 + mi * 16 + r + 8 * half;
                size_t o = (size_t)t * CH + c;
                out[o] = ybuf[o] + acc[mi][ni][r] + bv;
            }
        }
    }
}

// ---------------------------------------------------------------------------
extern "C" void kernel_launch(void* const* d_in, const int* in_sizes, int n_in,
                              void* d_out, int out_size, void* d_ws, size_t ws_size,
                              hipStream_t stream) {
    (void)in_sizes; (void)n_in; (void)out_size; (void)ws_size;
    const float* x      = (const float*)d_in[0];
    const float* g1     = (const float*)d_in[1];
    const float* be1    = (const float*)d_in[2];
    const float* w_qkv  = (const float*)d_in[3];
    const float* b_qkv  = (const float*)d_in[4];
    const float* rpt    = (const float*)d_in[5];
    const float* w_proj = (const float*)d_in[6];
    const float* b_proj = (const float*)d_in[7];
    const float* g2     = (const float*)d_in[8];
    const float* be2    = (const float*)d_in[9];
    const float* w_fc1  = (const float*)d_in[10];
    const float* b_fc1  = (const float*)d_in[11];
    const float* w_fc2  = (const float*)d_in[12];
    const float* b_fc2  = (const float*)d_in[13];
    float* out = (float*)d_out;

    char* ws = (char*)d_ws;
    size_t cur = 0;
    auto take = [&](size_t bytes) -> char* {
        char* p = ws + cur;
        cur += (bytes + 255) & ~(size_t)255;
        return p;
    };
    _Float16* wqkvT  = (_Float16*)take((size_t)768 * 256 * 2);
    _Float16* wprojT = (_Float16*)take((size_t)256 * 256 * 2);
    _Float16* wfc1T  = (_Float16*)take((size_t)1024 * 256 * 2);
    _Float16* wfc2T  = (_Float16*)take((size_t)256 * 1024 * 2);
    _Float16* h16    = (_Float16*)take((size_t)TOKENS * CH * 2);   // LN1 out; reused as LN2 out
    char* bigbase    = ws + cur;                                    // act16 aliases region below
    _Float16* q16    = (_Float16*)take((size_t)BNW * NH * 64 * 32 * 2);
    _Float16* k16    = (_Float16*)take((size_t)BNW * NH * 64 * 32 * 2);
    _Float16* vT16   = (_Float16*)take((size_t)BNW * NH * 32 * 64 * 2);
    _Float16* attnout= (_Float16*)take((size_t)TOKENS * CH * 2);
    _Float16* act16  = (_Float16*)bigbase;   // 50176*1024*2 fits in q/k/v/attn region (dead by FC1)
    float*    ybuf   = (float*)take((size_t)TOKENS * CH * 4);

    // weights -> f16 transposed
    cvt_wt_kernel<<<(768 * 256 + 255) / 256, 256, 0, stream>>>(w_qkv,  wqkvT,  256, 768);
    cvt_wt_kernel<<<(256 * 256 + 255) / 256, 256, 0, stream>>>(w_proj, wprojT, 256, 256);
    cvt_wt_kernel<<<(1024 * 256 + 255) / 256, 256, 0, stream>>>(w_fc1, wfc1T,  256, 1024);
    cvt_wt_kernel<<<(256 * 1024 + 255) / 256, 256, 0, stream>>>(w_fc2, wfc2T,  1024, 256);

    // LN1 + shift + window partition
    ln_win_kernel<<<TOKENS / 8, 256, 0, stream>>>(x, g1, be1, h16, 1);
    // zero pads, then QKV
    pad_zero_kernel<<<(8192 * 15 * 32 + 255) / 256, 256, 0, stream>>>(q16, k16, vT16);
    gemm_qkv_kernel<<<dim3(12, 196), 256, 0, stream>>>(h16, wqkvT, b_qkv, q16, k16, vT16);
    // windowed attention
    attn_kernel<<<BNW, 256, 0, stream>>>(q16, k16, vT16, rpt, attnout);
    // proj + reverse/unshift + residual
    gemm_proj_kernel<<<dim3(4, 196), 256, 0, stream>>>(attnout, wprojT, b_proj, x, ybuf);
    // MLP
    ln_win_kernel<<<TOKENS / 8, 256, 0, stream>>>(ybuf, g2, be2, h16, 0);
    gemm_fc1_kernel<<<dim3(16, 196), 256, 0, stream>>>(h16, wfc1T, b_fc1, act16);
    gemm_fc2_kernel<<<dim3(4, 196), 256, 0, stream>>>(act16, wfc2T, b_fc2, ybuf, out);
}
